// MultiHeadSelfAttention_65292092834051
// MI455X (gfx1250) — matrix operand order repaired and checked by reading, not compiled, not run on hardware
//
#include <hip/hip_runtime.h>
#include <stdint.h>

typedef _Float16 f16;
typedef __attribute__((ext_vector_type(16))) _Float16 v16h;
typedef __attribute__((ext_vector_type(8)))  float    v8f;

#define D_MODEL 1024
#define NUM_HEADS 16
#define DEPTH 64
#define SEQ 2048
#define BATCH 2
#define MTOT (BATCH * SEQ)   // 4096 rows total

// ---------------- CDNA5 async memory->LDS copy (guarded; falls back to sync)
#if __has_builtin(__builtin_amdgcn_global_load_async_to_lds_b128) && \
    __has_builtin(__builtin_amdgcn_s_wait_asynccnt)
#define HAVE_ASYNC 1
#else
#define HAVE_ASYNC 0
#endif

typedef int v4i __attribute__((vector_size(16)));            // matches builtin pointee
typedef __attribute__((address_space(1))) v4i gv4i;          // global (__device__) int4
typedef __attribute__((address_space(3))) v4i lv4i;          // LDS (__shared__) int4

__device__ __forceinline__ void cp16_to_lds(f16* ldst, const f16* gsrc) {
#if HAVE_ASYNC
    __builtin_amdgcn_global_load_async_to_lds_b128(
        (gv4i*)(unsigned long long)(uintptr_t)gsrc,
        (lv4i*)(unsigned)(uintptr_t)ldst, 0, 0);
#else
    *(uint4*)ldst = *(const uint4*)gsrc;
#endif
}
__device__ __forceinline__ void wait_cp() {
#if HAVE_ASYNC
    __builtin_amdgcn_s_wait_asynccnt(0);
#endif
}

// ---------------- DPP16 row (16-lane) reductions, result row-uniform, no LDS
template <int CTRL>
__device__ __forceinline__ float dpp_movf(float x) {
    return __int_as_float(
        __builtin_amdgcn_update_dpp(0, __float_as_int(x), CTRL, 0xF, 0xF, true));
}
__device__ __forceinline__ float row16_max(float x) {
    x = fmaxf(x, dpp_movf<0xB1>(x));   // quad_perm(1,0,3,2)  xor1
    x = fmaxf(x, dpp_movf<0x4E>(x));   // quad_perm(2,3,0,1)  xor2
    x = fmaxf(x, dpp_movf<0x141>(x));  // row_half_mirror     (crosses quads)
    x = fmaxf(x, dpp_movf<0x140>(x));  // row_mirror          (crosses 8s)
    return x;
}
__device__ __forceinline__ float row16_sum(float x) {
    x += dpp_movf<0xB1>(x);
    x += dpp_movf<0x4E>(x);
    x += dpp_movf<0x141>(x);
    x += dpp_movf<0x140>(x);
    return x;
}

// ---------------------------------------------------------------- cast f32->f16
__global__ void cast_f32_f16(const float* __restrict__ src, f16* __restrict__ dst, int n) {
    int i = blockIdx.x * 256 + threadIdx.x;
    if (i < n) dst[i] = (f16)src[i];
}

// ---------------------------------------------------------------- WMMA GEMM
// C[M,N] = A[M,K](f16) @ B[K,N](f16) + bias(f32). Tile 128x128, BK=32,
// 8 waves (2x4), each wave 64x32 = 4x2 WMMA tiles. Double-buffered LDS.
template <typename OUT>
__global__ __launch_bounds__(256)
void gemm_f16_wmma(const f16* __restrict__ A, const f16* __restrict__ B,
                   const float* __restrict__ bias, OUT* __restrict__ C,
                   int M, int N, int K) {
    const int LDA = 40;                    // halves per LDS row (32+8 pad), 80B stride
    __shared__ f16 As[2][128 * 40];        // [row][k]
    __shared__ f16 Bs[2][128 * 40];        // transposed: [n][k]

    const int tid  = threadIdx.x;
    const int lane = tid & 31;
    const int wave = tid >> 5;
    const int wm = wave >> 2, wn = wave & 3;
    const int bm = blockIdx.x * 128, bn = blockIdx.y * 128;
    const int hi = lane >> 4, ln = lane & 15;

    auto stageA = [&](int k0, int buf) {
        #pragma unroll
        for (int i = 0; i < 2; i++) {
            int idx = tid + i * 256;       // 0..511
            int row = idx >> 2, seg = idx & 3;
            cp16_to_lds(&As[buf][row * LDA + seg * 8],
                        A + (size_t)(bm + row) * K + k0 + seg * 8);
        }
    };
    auto stageB = [&](int k0, int buf) {
        #pragma unroll
        for (int i = 0; i < 2; i++) {
            int idx = tid + i * 256;
            int row = idx >> 4, seg = idx & 15;      // k-row, col segment
            uint4 d = *(const uint4*)(B + (size_t)(k0 + row) * N + bn + seg * 8);
            const f16* dh = (const f16*)&d;
            #pragma unroll
            for (int e = 0; e < 8; e++)
                Bs[buf][(seg * 8 + e) * LDA + row] = dh[e];
        }
    };

    v8f acc[4][2];
    #pragma unroll
    for (int i = 0; i < 4; i++)
        #pragma unroll
        for (int j = 0; j < 2; j++)
            #pragma unroll
            for (int e = 0; e < 8; e++) acc[i][j][e] = 0.0f;

    union F { v16h v; unsigned u[8]; };

    stageA(0, 0); stageB(0, 0);
    wait_cp();
    __syncthreads();

    int cur = 0;
    for (int k0 = 0; k0 < K; k0 += 32) {
        int nxt = cur ^ 1;
        if (k0 + 32 < K) {
            stageA(k0 + 32, nxt);
            stageB(k0 + 32, nxt);
            if (k0 + 64 < K)
                __builtin_prefetch(A + (size_t)(bm + (tid >> 2)) * K + k0 + 64, 0, 0);
        }

        const unsigned* Aw = (const unsigned*)As[cur];
        const unsigned* Bw = (const unsigned*)Bs[cur];
        F af[4], bf[2];
        // A fragments: 16-bit A 16x32 layout (ISA 7.12.2)
        #pragma unroll
        for (int mt = 0; mt < 4; mt++) {
            int base = (wm * 64 + mt * 16 + ln) * (LDA / 2);
            #pragma unroll
            for (int p = 0; p < 8; p++) {
                int c = (p & 3) + ((p >> 2) * 8) + hi * 4;   // u32 column
                af[mt].u[p] = Aw[base + c];
            }
        }
        // B fragments: same K order as A (k=8h+s for s<8, 16+8h+(s-8) for s>=8), N=lane%16
        #pragma unroll
        for (int nt = 0; nt < 2; nt++) {
            int base = (wn * 32 + nt * 16 + ln) * (LDA / 2);
            #pragma unroll
            for (int p = 0; p < 8; p++)
                bf[nt].u[p] = Bw[base + (p & 3) + ((p >> 2) * 8) + hi * 4];
        }
        #pragma unroll
        for (int mt = 0; mt < 4; mt++)
            #pragma unroll
            for (int nt = 0; nt < 2; nt++)
                acc[mt][nt] = __builtin_amdgcn_wmma_f32_16x16x32_f16(
                    false, af[mt].v, false, bf[nt].v, (short)0, acc[mt][nt], false, false);

        if (k0 + 32 < K) { wait_cp(); __syncthreads(); }
        cur = nxt;
    }

    // epilogue: bias + store (C layout: row = base + hi*8 + v, col = lane%16)
    #pragma unroll
    for (int mt = 0; mt < 4; mt++)
        #pragma unroll
        for (int nt = 0; nt < 2; nt++) {
            int colG = bn + wn * 32 + nt * 16 + ln;
            float bv = bias[colG];
            #pragma unroll
            for (int v = 0; v < 8; v++) {
                int rowG = bm + wm * 64 + mt * 16 + hi * 8 + v;
                C[(size_t)rowG * N + colG] = (OUT)(acc[mt][nt][v] + bv);
            }
        }
}

// ---------------------------------------------------------------- flash attention
// grid: (S/64, H, B); 128 threads = 4 waves; wave w owns query rows qt*64+w*16..+16.
// Double-buffered K/V tiles; 1/sqrt(d) folded into Q fragments.
__global__ __launch_bounds__(128)
void attention_wmma(const f16* __restrict__ Q, const f16* __restrict__ Kt,
                    const f16* __restrict__ Vt_g, f16* __restrict__ O) {
    const int qt = blockIdx.x, h = blockIdx.y, b = blockIdx.z;
    const int tid = threadIdx.x;
    const int lane = tid & 31, wave = tid >> 5;
    const int hi = lane >> 4, ln = lane & 15;

    const int LKV = 72;                 // halves per LDS row (64+8 pad) = 144B stride
    __shared__ f16 Ks[2][64 * 72];      // [key][d]
    __shared__ f16 Vs[2][64 * 72];      // transposed: [d][key]
    __shared__ f16 Ps[4][16 * 72];      // per-wave P scratch [row][key]

    const size_t headoff = (size_t)b * SEQ * D_MODEL + h * DEPTH;
    const int qbase = qt * 64 + wave * 16;

    union F { v16h v; unsigned u[8]; };

    auto stageKV = [&](int j, int buf) {
        #pragma unroll
        for (int i = 0; i < 4; i++) {
            int idx = tid + i * 128;        // 0..511
            int row = idx >> 3, seg = idx & 7;
            size_t g = headoff + (size_t)(j * 64 + row) * D_MODEL + seg * 8;
            cp16_to_lds(&Ks[buf][row * LKV + seg * 8], Kt + g);
            uint4 d = *(const uint4*)(Vt_g + g);
            const f16* dh = (const f16*)&d;
            #pragma unroll
            for (int e = 0; e < 8; e++)
                Vs[buf][(seg * 8 + e) * LKV + row] = dh[e];
        }
    };

    // Q fragments (2 K-chunks of 32 over depth=64), pre-scaled by 1/sqrt(64)
    F qf[2];
    {
        const unsigned* qw = (const unsigned*)(Q + headoff + (size_t)(qbase + ln) * D_MODEL);
        #pragma unroll
        for (int kc = 0; kc < 2; kc++) {
            #pragma unroll
            for (int p = 0; p < 8; p++) {
                int c = (p & 3) + ((p >> 2) * 8) + hi * 4 + kc * 16;
                qf[kc].u[p] = qw[c];
            }
            qf[kc].v = qf[kc].v * (f16)0.125f;   // exact pow2 scale
        }
    }

    float m[8], l[8];
    v8f acco[4];
    #pragma unroll
    for (int v = 0; v < 8; v++) { m[v] = -1e30f; l[v] = 0.0f; }
    #pragma unroll
    for (int dt = 0; dt < 4; dt++)
        #pragma unroll
        for (int v = 0; v < 8; v++) acco[dt][v] = 0.0f;

    const int NTILE = SEQ / 64;
    stageKV(0, 0);
    wait_cp();
    __syncthreads();

    int cur = 0;
    for (int j = 0; j < NTILE; j++) {
        int nxt = cur ^ 1;
        if (j + 1 < NTILE) stageKV(j + 1, nxt);

        // S = (Q/sqrt(d)) @ K^T   (B-frag: N = key, WMMA-K = depth)
        v8f sacc[4];
        const unsigned* Kw = (const unsigned*)Ks[cur];
        #pragma unroll
        for (int nt = 0; nt < 4; nt++) {
            #pragma unroll
            for (int e = 0; e < 8; e++) sacc[nt][e] = 0.0f;
            #pragma unroll
            for (int kc = 0; kc < 2; kc++) {
                F bf;
                int base = (nt * 16 + ln) * (LKV / 2);
                #pragma unroll
                for (int p = 0; p < 8; p++)
                    bf.u[p] = Kw[base + kc * 16 + (p & 3) + ((p >> 2) * 8) + hi * 4];
                sacc[nt] = __builtin_amdgcn_wmma_f32_16x16x32_f16(
                    false, qf[kc].v, false, bf.v, (short)0, sacc[nt], false, false);
            }
        }

        // online softmax (row stats uniform per 16-lane half; DPP reductions)
        #pragma unroll
        for (int v = 0; v < 8; v++) {
            float mx = fmaxf(fmaxf(sacc[0][v], sacc[1][v]),
                             fmaxf(sacc[2][v], sacc[3][v]));
            mx = row16_max(mx);
            float nm = fmaxf(m[v], mx);
            float alpha = __expf(m[v] - nm);
            m[v] = nm;
            float rs = 0.0f;
            #pragma unroll
            for (int nt = 0; nt < 4; nt++) {
                float p = __expf(sacc[nt][v] - nm);
                sacc[nt][v] = p;
                rs += p;
            }
            rs = row16_sum(rs);
            l[v] = l[v] * alpha + rs;
            #pragma unroll
            for (int dt = 0; dt < 4; dt++) acco[dt][v] *= alpha;
        }

        // P -> wave-private LDS (C layout), reload in A-fragment layout
        f16* myP = &Ps[wave][0];
        #pragma unroll
        for (int nt = 0; nt < 4; nt++)
            #pragma unroll
            for (int v = 0; v < 8; v++)
                myP[(hi * 8 + v) * LKV + nt * 16 + ln] = (f16)sacc[nt][v];
        asm volatile("" ::: "memory");   // keep store->load order (same-wave DS in-order)

        F pf[2];
        const unsigned* Pw = (const unsigned*)myP;
        #pragma unroll
        for (int kc = 0; kc < 2; kc++)
            #pragma unroll
            for (int p = 0; p < 8; p++) {
                int c = (p & 3) + ((p >> 2) * 8) + hi * 4 + kc * 16;
                pf[kc].u[p] = Pw[ln * (LKV / 2) + c];
            }

        // O += P @ V   (B-frag from depth-major Vs: N = depth col, WMMA-K = key)
        const unsigned* Vw = (const unsigned*)Vs[cur];
        #pragma unroll
        for (int dt = 0; dt < 4; dt++) {
            #pragma unroll
            for (int kc = 0; kc < 2; kc++) {
                F bf;
                int base = (dt * 16 + ln) * (LKV / 2);
                #pragma unroll
                for (int p = 0; p < 8; p++)
                    bf.u[p] = Vw[base + kc * 16 + (p & 3) + ((p >> 2) * 8) + hi * 4];
                acco[dt] = __builtin_amdgcn_wmma_f32_16x16x32_f16(
                    false, pf[kc].v, false, bf.v, (short)0, acco[dt], false, false);
            }
        }

        if (j + 1 < NTILE) { wait_cp(); __syncthreads(); }
        cur = nxt;
    }

    // normalize and store (f16) into attention-output buffer
    #pragma unroll
    for (int v = 0; v < 8; v++) {
        float inv = 1.0f / l[v];
        #pragma unroll
        for (int dt = 0; dt < 4; dt++) acco[dt][v] *= inv;
    }
    #pragma unroll
    for (int dt = 0; dt < 4; dt++)
        #pragma unroll
        for (int v = 0; v < 8; v++) {
            int row = qbase + hi * 8 + v;
            O[headoff + (size_t)row * D_MODEL + dt * 16 + ln] = (f16)acco[dt][v];
        }
}

// ---------------------------------------------------------------- launcher
extern "C" void kernel_launch(void* const* d_in, const int* in_sizes, int n_in,
                              void* d_out, int out_size, void* d_ws, size_t ws_size,
                              hipStream_t stream) {
    const float* X  = (const float*)d_in[0];
    const float* wq = (const float*)d_in[1];
    const float* bq = (const float*)d_in[2];
    const float* wk = (const float*)d_in[3];
    const float* bk = (const float*)d_in[4];
    const float* wv = (const float*)d_in[5];
    const float* bv = (const float*)d_in[6];
    const float* wo = (const float*)d_in[7];
    const float* bo = (const float*)d_in[8];
    float* out = (float*)d_out;

    const size_t NX = (size_t)MTOT * D_MODEL;       // 4M elems
    const size_t NW = (size_t)D_MODEL * D_MODEL;    // 1M elems

    f16* Xh  = (f16*)d_ws;
    f16* Wqh = Xh  + NX;
    f16* Wkh = Wqh + NW;
    f16* Wvh = Wkh + NW;
    f16* Woh = Wvh + NW;
    f16* Qh  = Woh + NW;
    f16* Kh  = Qh  + NX;
    f16* Vh  = Kh  + NX;
    f16* Ah  = Xh;   // alias: X no longer needed once Q/K/V exist

    cast_f32_f16<<<(int)((NX + 255) / 256), 256, 0, stream>>>(X,  Xh,  (int)NX);
    cast_f32_f16<<<(int)((NW + 255) / 256), 256, 0, stream>>>(wq, Wqh, (int)NW);
    cast_f32_f16<<<(int)((NW + 255) / 256), 256, 0, stream>>>(wk, Wkh, (int)NW);
    cast_f32_f16<<<(int)((NW + 255) / 256), 256, 0, stream>>>(wv, Wvh, (int)NW);
    cast_f32_f16<<<(int)((NW + 255) / 256), 256, 0, stream>>>(wo, Woh, (int)NW);

    dim3 gg(MTOT / 128, D_MODEL / 128);
    gemm_f16_wmma<f16><<<gg, 256, 0, stream>>>(Xh, Wqh, bq, Qh, MTOT, D_MODEL, D_MODEL);
    gemm_f16_wmma<f16><<<gg, 256, 0, stream>>>(Xh, Wkh, bk, Kh, MTOT, D_MODEL, D_MODEL);
    gemm_f16_wmma<f16><<<gg, 256, 0, stream>>>(Xh, Wvh, bv, Vh, MTOT, D_MODEL, D_MODEL);

    dim3 ga(SEQ / 64, NUM_HEADS, BATCH);
    attention_wmma<<<ga, 128, 0, stream>>>(Qh, Kh, Vh, Ah);

    gemm_f16_wmma<float><<<gg, 256, 0, stream>>>(Ah, Woh, bo, out, MTOT, D_MODEL, D_MODEL);
}
